// UniversalScaleAdapter_44135083933751
// MI455X (gfx1250) — matrix operand order repaired and checked by reading, not compiled, not run on hardware
//
#include <hip/hip_runtime.h>
#include <cstdint>
#include <cstddef>

// Problem constants (match reference)
#define B_   2
#define NQ_  512
#define NC_  2048
#define D_   768
#define H_   12
#define HD_  64
#define FF_  3072
#define RB_  64

typedef _Float16 half_t;
typedef __attribute__((ext_vector_type(16))) _Float16 v16h;
typedef __attribute__((ext_vector_type(8)))  _Float16 v8h;
typedef __attribute__((ext_vector_type(8)))  float    v8f;
typedef unsigned int v4u __attribute__((ext_vector_type(4)));
typedef int          v8i __attribute__((ext_vector_type(8)));
typedef int          v4i __attribute__((ext_vector_type(4)));

#ifndef USE_TDM
#define USE_TDM 1
#endif

#define WMMA_F16(a, b, c) \
  __builtin_amdgcn_wmma_f32_16x16x32_f16(false, (a), false, (b), (short)0, (c), false, false)

__device__ __forceinline__ float gelu_exact(float x) {
  return 0.5f * x * (1.0f + erff(x * 0.70710678118654752f));
}

// ---- WMMA fragment loaders (wave32 layouts per CDNA5 ISA) ----
// A-matrix 16x32 f16: lane l: m=l&15, kh=l>>4; vgpr pair vg holds
//   k = kh*16 + vg*2 (+0/1)  -> contiguous (matches frag_bt_ld)
__device__ __forceinline__ v16h frag_a_ld(const half_t* src, int ld, int lane) {
  const int m = lane & 15, kh = lane >> 4;
  v16h a;
#pragma unroll
  for (int vg = 0; vg < 8; ++vg) {
    const int k = kh * 16 + (vg << 1);
    a[2 * vg]     = src[m * ld + k];
    a[2 * vg + 1] = src[m * ld + k + 1];
  }
  return a;
}
// B-matrix 32x16, source stored K-contiguous per output column ([n][k] row-major):
//   lane l: n=l&15, kh=l>>4; b[e] = src[n*ld + kh*16 + e]  (two b128 ds loads when
//   ld*2 bytes is a multiple of 16 -> use ld = 72)
__device__ __forceinline__ v16h frag_bt_ld(const half_t* src, int ld, int lane) {
  const int n = lane & 15, kh = lane >> 4;
  v16h b;
#pragma unroll
  for (int e = 0; e < 16; ++e) b[e] = src[n * ld + kh * 16 + e];
  return b;
}

#if USE_TDM
// TDM: stage a 64(n-rows) x 64(k) fp16 tile of Wt (N x K row-major) into LDS,
// with 4 dwords of padding per 32-dword row (-> LDS ld = 72 halves, 144B:
// 16B-aligned rows so fragments lower to ds_load_b128, 36-dword row stride for banks).
__device__ __forceinline__ void tdm_load_wtile(const half_t* Wt, int K, int Ntot,
                                               int nrow0, int kt, half_t* ldsdst) {
  const uint64_t gaddr = (uint64_t)(uintptr_t)(Wt + (size_t)nrow0 * K + kt);
  v4u g0;
  g0[0] = 1u;                                    // count=1 valid descriptor
  g0[1] = (uint32_t)(uintptr_t)ldsdst;           // lds_addr (byte offset)
  g0[2] = (uint32_t)gaddr;                       // global_addr[31:0]
  g0[3] = ((uint32_t)(gaddr >> 32) & 0x01FFFFFFu) | (2u << 30);  // addr[56:32] | type=2
  const uint32_t td0 = (uint32_t)K;              // tensor_dim0 (row width, elems)
  const uint32_t td1 = (uint32_t)Ntot;           // tensor_dim1 (#rows)
  const uint64_t st0 = (uint64_t)(uint32_t)K;    // dim0 stride (elems)
  v8i g1;
  g1[0] = (int)((1u << 16) |                     // data_size = 2B (fp16)
                (1u << 20) |                     // pad_enable
                (4u << 22) |                     // pad_interval = 32 dwords (one row)
                (3u << 25));                     // pad_amount = 4 dwords
  g1[1] = (int)(td0 << 16);                      // tensor_dim0[15:0] (atomic addr = 0)
  g1[2] = (int)((td0 >> 16) | ((td1 & 0xFFFFu) << 16));
  g1[3] = (int)(((td1 >> 16) & 0xFFFFu) | (64u << 16));  // tile_dim0 = 64 (k)
  g1[4] = 64;                                    // tile_dim1 = 64 (n), tile_dim2 = 0
  g1[5] = (int)(uint32_t)st0;
  g1[6] = (int)((uint32_t)(st0 >> 32) & 0xFFFFu);
  g1[7] = 0;
  v4i gz = {0, 0, 0, 0};
#if defined(__clang_major__) && (__clang_major__ >= 23)
  v8i gz8 = {0, 0, 0, 0, 0, 0, 0, 0};
  __builtin_amdgcn_tensor_load_to_lds(g0, g1, gz, gz, gz8, 0);
#else
  __builtin_amdgcn_tensor_load_to_lds(g0, g1, gz, gz, 0);
#endif
}
#endif

// one b128 global load + one b128 LDS store per thread (32x64 tile, 256 threads)
__device__ __forceinline__ void stage_a_tile(const half_t* A, half_t (*As)[72],
                                             size_t brow, int K, int kt, int tid) {
  const int row = tid >> 3;
  const int col = (tid & 7) * 8;
  *(v8h*)(&As[row][col]) = *(const v8h*)(A + (brow + row) * (size_t)K + kt + col);
}

// =====================================================================
// Tiled WMMA GEMM:  out = [residual +] [gelu] (A(MxK) @ W(KxN) + bias)
// W passed pre-transposed as Wt (N x K). Block: 256 threads = 8 waves,
// C-tile 32x64, 1 frag/wave, BK=64 (2 wmma/iter), double-buffered LDS:
// TDM DMA for tile t+1 overlaps the WMMAs of tile t.
// =====================================================================
__global__ __launch_bounds__(256) void gemm_wmma_kernel(
    const half_t* __restrict__ A, const half_t* __restrict__ Wt,
    const float* __restrict__ bias, const float* __restrict__ residual,
    float* __restrict__ outF, half_t* __restrict__ outH,
    int M, int N, int K, int act_gelu) {
  __shared__ half_t As[2][32][72];
  __shared__ half_t Bs[2][64 * 72];   // [n][k], ld=72 (TDM 4-dword row padding)
  const int tid  = threadIdx.x;
  const int lane = tid & 31;
  const int wave = tid >> 5;
  const int msub = wave >> 2, nsub = wave & 3;
  const size_t brow = (size_t)blockIdx.y * 32;
  const int bcol = blockIdx.x * 64;
  const int nt = K / 64;

  v8f acc = {0.f, 0.f, 0.f, 0.f, 0.f, 0.f, 0.f, 0.f};

  // prologue: tile 0
#if USE_TDM
  if (tid < 32) tdm_load_wtile(Wt, K, N, bcol, 0, &Bs[0][0]);
#else
  for (int i = tid; i < 64 * 64; i += 256)
    Bs[0][(i >> 6) * 72 + (i & 63)] = Wt[(size_t)(bcol + (i >> 6)) * K + (i & 63)];
#endif
  stage_a_tile(A, As[0], brow, K, 0, tid);
#if USE_TDM
  __builtin_amdgcn_s_wait_tensorcnt(0);
#endif
  __syncthreads();

  int cur = 0;
  for (int t = 0; t < nt; ++t) {
    const int nxt = cur ^ 1;
    if (t + 1 < nt) {
#if USE_TDM
      if (tid < 32) tdm_load_wtile(Wt, K, N, bcol, (t + 1) * 64, &Bs[nxt][0]);
#else
      for (int i = tid; i < 64 * 64; i += 256)
        Bs[nxt][(i >> 6) * 72 + (i & 63)] =
            Wt[(size_t)(bcol + (i >> 6)) * K + (t + 1) * 64 + (i & 63)];
#endif
      stage_a_tile(A, As[nxt], brow, K, (t + 1) * 64, tid);
    }
    // compute on current tile while the DMA for the next tile is in flight
#pragma unroll
    for (int st = 0; st < 2; ++st) {
      v16h af = frag_a_ld(&As[cur][msub * 16][st * 32], 72, lane);
      v16h bf = frag_bt_ld(&Bs[cur][(nsub * 16) * 72 + st * 32], 72, lane);
      acc = WMMA_F16(af, bf, acc);
    }
#if USE_TDM
    __builtin_amdgcn_s_wait_tensorcnt(0);   // no-op for non-issuing waves
#endif
    __syncthreads();
    cur = nxt;
  }

  // Epilogue: C layout -> element vg: row = vg + 8*(lane>=16), col = lane&15
  const int col = bcol + nsub * 16 + (lane & 15);
  const int kh  = lane >> 4;
#pragma unroll
  for (int vg = 0; vg < 8; ++vg) {
    const size_t row = brow + msub * 16 + vg + 8 * kh;
    float v = acc[vg];
    if (bias)     v += bias[col];
    if (act_gelu) v  = gelu_exact(v);
    if (residual) v += residual[row * N + col];
    if (outF) outF[row * N + col] = v;
    if (outH) outH[row * N + col] = (half_t)v;
  }
}

// =====================================================================
// Fused flash-style attention (self or cross+bias).
// Block: 128 threads = 4 waves; 16-query tile; K chunk kept row-major
// (K^T frags contiguous b128), V chunk staged TRANSPOSED so P@V frags are
// contiguous b128 ds loads. Softmax parallelized 8 threads/row.
// =====================================================================
__global__ __launch_bounds__(128) void attn_wmma_kernel(
    const half_t* __restrict__ Qp, const half_t* __restrict__ Kp,
    const half_t* __restrict__ Vp, const half_t* __restrict__ biasp,
    half_t* __restrict__ Op,
    int nk, int ldq, int ldkv, int qoff, int koff, int voff) {
  __shared__ half_t Qs[16][64];
  __shared__ half_t Ks[64][72];      // [key][d], 144B rows (b128-aligned)
  __shared__ half_t Vt[64][72];      // [d][key]  (transposed stage)
  __shared__ float  Ss[16][64];
  __shared__ half_t Ps[16][64];
  __shared__ float  rowM[16], rowL[16], rowA[16];
  __shared__ float  red2[16][8];

  const int tid = threadIdx.x, lane = tid & 31, wave = tid >> 5;
  const int b = blockIdx.z, h = blockIdx.y, qt = blockIdx.x;
  const int colL = lane & 15, kh = lane >> 4;
  const int sr = tid >> 3, sc = tid & 7;   // softmax: 8 threads per row
  const float scale = 0.125f;              // 1/sqrt(64)

  const size_t qbase = ((size_t)b * NQ_ + qt * 16) * (size_t)ldq + h * HD_ + qoff;
  const size_t kbase = (size_t)b * nk * (size_t)ldkv + h * HD_ + koff;
  const size_t vbase = (size_t)b * nk * (size_t)ldkv + h * HD_ + voff;

  for (int i = tid; i < 16 * 64; i += 128)
    Qs[i >> 6][i & 63] = Qp[qbase + (size_t)(i >> 6) * ldq + (i & 63)];
  if (tid < 16) { rowM[tid] = -3.0e38f; rowL[tid] = 0.f; }

  v8f oacc = {0.f, 0.f, 0.f, 0.f, 0.f, 0.f, 0.f, 0.f};

  for (int kb = 0; kb < nk; kb += 64) {
    __syncthreads();
    for (int i = tid; i < 64 * 64; i += 128) {
      const int r = i >> 6, c = i & 63;
      Ks[r][c] = Kp[kbase + (size_t)(kb + r) * ldkv + c];
      Vt[c][r] = Vp[vbase + (size_t)(kb + r) * ldkv + c];
    }
    if (kb + 64 < nk)  // speculative prefetch of next K chunk -> global_prefetch_b8
      __builtin_prefetch(Kp + kbase + (size_t)(kb + 64) * ldkv, 0, 1);
    __syncthreads();

    // S frag: this wave owns keys [wave*16, wave*16+16)
    v8f s = {0.f, 0.f, 0.f, 0.f, 0.f, 0.f, 0.f, 0.f};
#pragma unroll
    for (int st = 0; st < 2; ++st) {
      v16h af = frag_a_ld(&Qs[0][st * 32], 64, lane);
      v16h bf = frag_bt_ld(&Ks[wave * 16][st * 32], 72, lane);
      s = WMMA_F16(af, bf, s);
    }
#pragma unroll
    for (int vg = 0; vg < 8; ++vg) {
      const int r = vg + 8 * kh;
      float lg = s[vg] * scale;
      if (biasp)
        lg += (float)biasp[(((size_t)b * H_ + h) * NQ_ + qt * 16 + r) * (size_t)nk +
                           kb + wave * 16 + colL];
      Ss[r][wave * 16 + colL] = lg;
    }
    __syncthreads();

    // --- online softmax, 8 threads per row ---
    {
      float pm = -3.0e38f;
#pragma unroll
      for (int c = 0; c < 8; ++c) pm = fmaxf(pm, Ss[sr][sc * 8 + c]);
      red2[sr][sc] = pm;
    }
    __syncthreads();
    if (tid < 16) {
      float mnew = rowM[tid];
#pragma unroll
      for (int j = 0; j < 8; ++j) mnew = fmaxf(mnew, red2[tid][j]);
      rowA[tid] = __expf(rowM[tid] - mnew);
      rowM[tid] = mnew;
    }
    __syncthreads();
    {
      const float mr = rowM[sr];
      float psum = 0.f;
#pragma unroll
      for (int c = 0; c < 8; ++c) {
        const float p = __expf(Ss[sr][sc * 8 + c] - mr);
        Ps[sr][sc * 8 + c] = (half_t)p;
        psum += p;
      }
      red2[sr][sc] = psum;
    }
    __syncthreads();
    if (tid < 16) {
      float sum = 0.f;
#pragma unroll
      for (int j = 0; j < 8; ++j) sum += red2[tid][j];
      rowL[tid] = rowL[tid] * rowA[tid] + sum;
    }
    __syncthreads();

    // Rescale O accumulator and add P @ V_chunk (this wave owns d-tile = wave)
#pragma unroll
    for (int vg = 0; vg < 8; ++vg) oacc[vg] *= rowA[vg + 8 * kh];
#pragma unroll
    for (int st = 0; st < 2; ++st) {
      v16h af = frag_a_ld(&Ps[0][st * 32], 64, lane);
      v16h bf = frag_bt_ld(&Vt[wave * 16][st * 32], 72, lane);
      oacc = WMMA_F16(af, bf, oacc);
    }
  }
  __syncthreads();

#pragma unroll
  for (int vg = 0; vg < 8; ++vg) {
    const int r = vg + 8 * kh;
    const float o = oacc[vg] / rowL[r];
    Op[((size_t)b * NQ_ + qt * 16 + r) * (size_t)D_ + h * HD_ + wave * 16 + colL] =
        (half_t)o;
  }
}

// ---------------- LayerNorm (fp32 in -> fp16 out), one row per block ----------------
__global__ __launch_bounds__(256) void ln_kernel(const float* __restrict__ x,
                                                 const float* __restrict__ g,
                                                 const float* __restrict__ bta,
                                                 half_t* __restrict__ out, int D) {
  __shared__ float red[256];
  const int tid = threadIdx.x;
  const size_t base = (size_t)blockIdx.x * D;
  float s = 0.f;
  for (int i = tid; i < D; i += 256) s += x[base + i];
  red[tid] = s; __syncthreads();
  for (int o = 128; o > 0; o >>= 1) { if (tid < o) red[tid] += red[tid + o]; __syncthreads(); }
  const float mean = red[0] / D;
  __syncthreads();
  s = 0.f;
  for (int i = tid; i < D; i += 256) { const float d = x[base + i] - mean; s += d * d; }
  red[tid] = s; __syncthreads();
  for (int o = 128; o > 0; o >>= 1) { if (tid < o) red[tid] += red[tid + o]; __syncthreads(); }
  const float rstd = rsqrtf(red[0] / D + 1e-5f);
  for (int i = tid; i < D; i += 256)
    out[base + i] = (half_t)((x[base + i] - mean) * rstd * g[i] + bta[i]);
}

// ---------------- fp32 -> fp16 convert ----------------
__global__ void cvt_f32_f16_kernel(const float* __restrict__ in,
                                   half_t* __restrict__ out, size_t n) {
  size_t i = (size_t)blockIdx.x * blockDim.x + threadIdx.x;
  const size_t stride = (size_t)gridDim.x * blockDim.x;
  for (; i < n; i += stride) out[i] = (half_t)in[i];
}

// ------------- fp32 (RxC) -> fp16 transposed (CxR): weights become K-contiguous -------------
__global__ __launch_bounds__(256) void transpose_f32_f16_kernel(
    const float* __restrict__ in, half_t* __restrict__ out, int R, int C) {
  __shared__ float tile[32][33];
  const int tid = threadIdx.x;
  const int tx = tid & 31, ty = tid >> 5;
  const int r0 = blockIdx.y * 32, c0 = blockIdx.x * 32;
  for (int rr = ty; rr < 32; rr += 8)
    tile[rr][tx] = in[(size_t)(r0 + rr) * C + c0 + tx];
  __syncthreads();
  for (int cc = ty; cc < 32; cc += 8)
    out[(size_t)(c0 + cc) * R + r0 + tx] = (half_t)tile[tx][cc];
}

// ---------------- coordinate projection: P = coords @ rb_w1 (+ rb_b1) ----------------
// Exploits linearity: delta@W1 = qc@W1 - cc@W1.
__global__ void coordproj_kernel(const float* __restrict__ coords,
                                 const float* __restrict__ w1,
                                 const float* __restrict__ b1,
                                 float* __restrict__ P, int rows, int add_b) {
  const int total = rows * RB_;
  for (int i = blockIdx.x * blockDim.x + threadIdx.x; i < total;
       i += gridDim.x * blockDim.x) {
    const int n = i >> 6, j = i & 63;
    float v = coords[2 * n] * w1[j] + coords[2 * n + 1] * w1[RB_ + j];
    if (add_b) v += b1[j];
    P[i] = v;
  }
}

// ------------- bias tensor: bias[b,h,q,c] = gelu(P1[q]-P2[c]) @ rb_w2 + rb_b2 -------------
__global__ __launch_bounds__(256) void biasmlp_kernel(
    const float* __restrict__ P1, const float* __restrict__ P2,
    const float* __restrict__ w2, const float* __restrict__ b2,
    half_t* __restrict__ biasO) {
  __shared__ float p1s[RB_];
  __shared__ float w2s[RB_ * H_];
  __shared__ float b2s[H_];
  const int tid = threadIdx.x;
  const int bb = blockIdx.z, q = blockIdx.y;
  const int c = blockIdx.x * 256 + tid;
  if (tid < RB_) p1s[tid] = P1[((size_t)bb * NQ_ + q) * RB_ + tid];
  for (int i = tid; i < RB_ * H_; i += 256) w2s[i] = w2[i];
  if (tid < H_) b2s[tid] = b2[tid];
  __syncthreads();
  const float* p2 = P2 + ((size_t)bb * NC_ + c) * RB_;
  float acc[H_];
#pragma unroll
  for (int t = 0; t < H_; ++t) acc[t] = b2s[t];
  for (int j = 0; j < RB_; ++j) {
    const float hc = gelu_exact(p1s[j] - p2[j]);
#pragma unroll
    for (int t = 0; t < H_; ++t) acc[t] += hc * w2s[j * H_ + t];
  }
#pragma unroll
  for (int t = 0; t < H_; ++t)
    biasO[(((size_t)bb * H_ + t) * NQ_ + q) * (size_t)NC_ + c] = (half_t)acc[t];
}

// =====================================================================
extern "C" void kernel_launch(void* const* d_in, const int* in_sizes, int n_in,
                              void* d_out, int out_size, void* d_ws, size_t ws_size,
                              hipStream_t stream) {
  (void)in_sizes; (void)n_in; (void)out_size; (void)ws_size;

  const float* query_state    = (const float*)d_in[0];
  const float* context_k      = (const float*)d_in[1];
  const float* context_v      = (const float*)d_in[2];
  const float* query_coords   = (const float*)d_in[3];
  const float* context_coords = (const float*)d_in[4];
  const float* sn_g = (const float*)d_in[5];
  const float* sn_b = (const float*)d_in[6];
  const float* wqkv = (const float*)d_in[7];
  const float* bqkv = (const float*)d_in[8];
  const float* wo_s = (const float*)d_in[9];
  const float* bo_s = (const float*)d_in[10];
  const float* cn_g = (const float*)d_in[11];
  const float* cn_b = (const float*)d_in[12];
  const float* wq   = (const float*)d_in[13];
  const float* bq   = (const float*)d_in[14];
  const float* wk   = (const float*)d_in[15];
  const float* bk   = (const float*)d_in[16];
  const float* wv   = (const float*)d_in[17];
  const float* bv   = (const float*)d_in[18];
  const float* wo_c = (const float*)d_in[19];
  const float* bo_c = (const float*)d_in[20];
  const float* rb_w1 = (const float*)d_in[21];
  const float* rb_b1 = (const float*)d_in[22];
  const float* rb_w2 = (const float*)d_in[23];
  const float* rb_b2 = (const float*)d_in[24];
  const float* fn_g = (const float*)d_in[25];
  const float* fn_b = (const float*)d_in[26];
  const float* fw1  = (const float*)d_in[27];
  const float* fb1  = (const float*)d_in[28];
  const float* fw2  = (const float*)d_in[29];
  const float* fb2  = (const float*)d_in[30];

  // ---- workspace carve-out (~115 MB; bias tensor dominates, fits in 192MB L2) ----
  char* ws = (char*)d_ws;
  size_t cur = 0;
  auto alloc = [&](size_t elems, size_t esz) -> void* {
    void* p = ws + cur;
    cur += (elems * esz + 255) & ~(size_t)255;
    return p;
  };
  // transposed fp16 weights (N x K layout)
  half_t* wqkv_t = (half_t*)alloc((size_t)D_ * 3 * D_, 2);
  half_t* wos_t  = (half_t*)alloc((size_t)D_ * D_, 2);
  half_t* wq_t   = (half_t*)alloc((size_t)D_ * D_, 2);
  half_t* wk_t   = (half_t*)alloc((size_t)D_ * D_, 2);
  half_t* wv_t   = (half_t*)alloc((size_t)D_ * D_, 2);
  half_t* woc_t  = (half_t*)alloc((size_t)D_ * D_, 2);
  half_t* fw1_t  = (half_t*)alloc((size_t)D_ * FF_, 2);
  half_t* fw2_t  = (half_t*)alloc((size_t)FF_ * D_, 2);
  half_t* ck_h    = (half_t*)alloc((size_t)B_ * NC_ * D_, 2);
  half_t* cv_h    = (half_t*)alloc((size_t)B_ * NC_ * D_, 2);
  half_t* xn_h    = (half_t*)alloc((size_t)B_ * NQ_ * D_, 2);
  half_t* qkv_h   = (half_t*)alloc((size_t)B_ * NQ_ * 3 * D_, 2);
  half_t* o_h     = (half_t*)alloc((size_t)B_ * NQ_ * D_, 2);
  half_t* q_h     = (half_t*)alloc((size_t)B_ * NQ_ * D_, 2);
  half_t* kproj_h = (half_t*)alloc((size_t)B_ * NC_ * D_, 2);
  half_t* vproj_h = (half_t*)alloc((size_t)B_ * NC_ * D_, 2);
  half_t* ff_h    = (half_t*)alloc((size_t)B_ * NQ_ * FF_, 2);
  half_t* bias_h  = (half_t*)alloc((size_t)B_ * H_ * NQ_ * NC_, 2);
  float*  xres    = (float*)alloc((size_t)B_ * NQ_ * D_, 4);
  float*  P1      = (float*)alloc((size_t)B_ * NQ_ * RB_, 4);
  float*  P2      = (float*)alloc((size_t)B_ * NC_ * RB_, 4);

  auto cvt = [&](const float* src, half_t* dst, size_t n) {
    int blocks = (int)((n + 255) / 256);
    if (blocks > 2048) blocks = 2048;
    cvt_f32_f16_kernel<<<dim3(blocks), dim3(256), 0, stream>>>(src, dst, n);
  };
  auto tpose = [&](const float* src, half_t* dst, int R, int C) {
    transpose_f32_f16_kernel<<<dim3(C / 32, R / 32), dim3(256), 0, stream>>>(src, dst, R, C);
  };
  auto gemm = [&](const half_t* A, const half_t* Wt, const float* bias,
                  const float* resid, float* oF, half_t* oH,
                  int M, int N, int K, int act) {
    gemm_wmma_kernel<<<dim3(N / 64, M / 32), dim3(256), 0, stream>>>(
        A, Wt, bias, resid, oF, oH, M, N, K, act);
  };

  // 0) fp16 staging: weights transposed to (N x K), context tensors converted
  tpose(wqkv, wqkv_t, D_, 3 * D_);
  tpose(wo_s, wos_t, D_, D_);
  tpose(wq, wq_t, D_, D_);
  tpose(wk, wk_t, D_, D_);
  tpose(wv, wv_t, D_, D_);
  tpose(wo_c, woc_t, D_, D_);
  tpose(fw1, fw1_t, D_, FF_);
  tpose(fw2, fw2_t, FF_, D_);
  cvt(context_k, ck_h, (size_t)B_ * NC_ * D_);
  cvt(context_v, cv_h, (size_t)B_ * NC_ * D_);

  // 1) pre-norm self-attention
  ln_kernel<<<dim3(B_ * NQ_), dim3(256), 0, stream>>>(query_state, sn_g, sn_b, xn_h, D_);
  gemm(xn_h, wqkv_t, bqkv, nullptr, nullptr, qkv_h, B_ * NQ_, 3 * D_, D_, 0);
  attn_wmma_kernel<<<dim3(NQ_ / 16, H_, B_), dim3(128), 0, stream>>>(
      qkv_h, qkv_h, qkv_h, nullptr, o_h, NQ_, 3 * D_, 3 * D_, 0, D_, 2 * D_);
  gemm(o_h, wos_t, bo_s, query_state, xres, nullptr, B_ * NQ_, D_, D_, 0);

  // 2) relative-coordinate bias (linearity trick: delta@W1 = qc@W1 - cc@W1)
  coordproj_kernel<<<dim3(64), dim3(256), 0, stream>>>(query_coords, rb_w1, rb_b1,
                                                       P1, B_ * NQ_, 1);
  coordproj_kernel<<<dim3(256), dim3(256), 0, stream>>>(context_coords, rb_w1, rb_b1,
                                                        P2, B_ * NC_, 0);
  biasmlp_kernel<<<dim3(NC_ / 256, NQ_, B_), dim3(256), 0, stream>>>(
      P1, P2, rb_w2, rb_b2, bias_h);

  // 3) pre-norm cross-attention with bias
  ln_kernel<<<dim3(B_ * NQ_), dim3(256), 0, stream>>>(xres, cn_g, cn_b, xn_h, D_);
  gemm(xn_h, wq_t, bq, nullptr, nullptr, q_h, B_ * NQ_, D_, D_, 0);
  gemm(ck_h, wk_t, bk, nullptr, nullptr, kproj_h, B_ * NC_, D_, D_, 0);
  gemm(cv_h, wv_t, bv, nullptr, nullptr, vproj_h, B_ * NC_, D_, D_, 0);
  attn_wmma_kernel<<<dim3(NQ_ / 16, H_, B_), dim3(128), 0, stream>>>(
      q_h, kproj_h, vproj_h, bias_h, o_h, NC_, D_, D_, 0, 0, 0);
  gemm(o_h, woc_t, bo_c, xres, xres, nullptr, B_ * NQ_, D_, D_, 0);

  // 4) pre-norm FFN (gelu fused into first GEMM, residual into second)
  ln_kernel<<<dim3(B_ * NQ_), dim3(256), 0, stream>>>(xres, fn_g, fn_b, xn_h, D_);
  gemm(xn_h, fw1_t, fb1, nullptr, nullptr, ff_h, B_ * NQ_, FF_, D_, 1);
  gemm(ff_h, fw2_t, fb2, xres, (float*)d_out, nullptr, B_ * NQ_, D_, FF_, 0);
}
